// PointnetSAModule_72052371357927
// MI455X (gfx1250) — matrix operand order repaired and checked by reading, not compiled, not run on hardware
//
#include <hip/hip_runtime.h>

typedef __attribute__((ext_vector_type(16))) _Float16 v16h;
typedef __attribute__((ext_vector_type(8)))  _Float16 v8h;
typedef __attribute__((ext_vector_type(8)))  float    v8f;

#define NB        8
#define NPTS      8192
#define NPOINT    2048
#define NSAMPLE   32
#define ROWS      (NB*NPOINT*NSAMPLE)   // 524288
#define R2        0.16f
#define BN_CNT_INV (1.0f/524288.0f)

// ---------------------------------------------------------------------------
// 1) Furthest point sampling: one block per batch, dists kept in LDS.
//    Matches reference: idx[0]=0, then 2047 argmax steps (first-max index).
// ---------------------------------------------------------------------------
__global__ void fps_kernel(const float* __restrict__ xyz,
                           int* __restrict__ fps_idx,
                           float* __restrict__ newxyz)
{
    __shared__ float dists[NPTS];
    __shared__ float rv[32];
    __shared__ int   ri[32];
    __shared__ int   s_last;
    __shared__ float s_p[3];

    const int b   = blockIdx.x;
    const int tid = threadIdx.x;           // 1024 threads = 32 waves (wave32)
    const float* X = xyz + (size_t)b * NPTS * 3;

    for (int i = tid; i < NPTS; i += 1024) dists[i] = 1e10f;
    if (tid == 0) {
        fps_idx[b * NPOINT] = 0;
        newxyz[((size_t)b * NPOINT) * 3 + 0] = X[0];
        newxyz[((size_t)b * NPOINT) * 3 + 1] = X[1];
        newxyz[((size_t)b * NPOINT) * 3 + 2] = X[2];
        s_last = 0;
    }
    __syncthreads();

    int last = 0;
    for (int it = 1; it < NPOINT; ++it) {
        if (tid == 0) {
            s_p[0] = X[last * 3 + 0];
            s_p[1] = X[last * 3 + 1];
            s_p[2] = X[last * 3 + 2];
        }
        __syncthreads();
        const float px = s_p[0], py = s_p[1], pz = s_p[2];

        float bv = -1.0f; int bi = NPTS;
        for (int i = tid; i < NPTS; i += 1024) {
            float dx = X[i * 3 + 0] - px;
            float dy = X[i * 3 + 1] - py;
            float dz = X[i * 3 + 2] - pz;
            float d2 = dx * dx + dy * dy + dz * dz;
            float dc = dists[i];
            if (d2 < dc) { dc = d2; dists[i] = d2; }
            if (dc > bv || (dc == bv && i < bi)) { bv = dc; bi = i; }
        }
        // wave32 reduction, first-max semantics (smaller index wins ties)
        #pragma unroll
        for (int off = 16; off; off >>= 1) {
            float ov = __shfl_xor(bv, off, 32);
            int   oi = __shfl_xor(bi, off, 32);
            if (ov > bv || (ov == bv && oi < bi)) { bv = ov; bi = oi; }
        }
        const int lane = tid & 31, w = tid >> 5;
        if (lane == 0) { rv[w] = bv; ri[w] = bi; }
        __syncthreads();
        if (w == 0) {
            bv = rv[lane]; bi = ri[lane];
            #pragma unroll
            for (int off = 16; off; off >>= 1) {
                float ov = __shfl_xor(bv, off, 32);
                int   oi = __shfl_xor(bi, off, 32);
                if (ov > bv || (ov == bv && oi < bi)) { bv = ov; bi = oi; }
            }
            if (lane == 0) {
                s_last = bi;
                fps_idx[b * NPOINT + it] = bi;
                newxyz[((size_t)b * NPOINT + it) * 3 + 0] = X[bi * 3 + 0];
                newxyz[((size_t)b * NPOINT + it) * 3 + 1] = X[bi * 3 + 1];
                newxyz[((size_t)b * NPOINT + it) * 3 + 2] = X[bi * 3 + 2];
            }
        }
        __syncthreads();
        last = s_last;
    }
}

// ---------------------------------------------------------------------------
// 2) Ball query: one wave32 per center; ballot+prefix-popc emits the
//    32 smallest in-radius indices in ascending order (== reference sort).
// ---------------------------------------------------------------------------
__global__ void ball_query_kernel(const float* __restrict__ xyz,
                                  const float* __restrict__ newxyz,
                                  int* __restrict__ idxb)
{
    const int c    = blockIdx.x * 8 + (threadIdx.x >> 5);  // 16384 centers
    const int lane = threadIdx.x & 31;
    const int b    = c >> 11;
    const float* X = xyz + (size_t)b * NPTS * 3;
    const float qx = newxyz[(size_t)c * 3 + 0];
    const float qy = newxyz[(size_t)c * 3 + 1];
    const float qz = newxyz[(size_t)c * 3 + 2];
    int* out = idxb + (size_t)c * NSAMPLE;

    int cnt = 0, first = 0; bool found = false;
    for (int n0 = 0; n0 < NPTS; n0 += 32) {
        const int i = n0 + lane;
        float dx = X[i * 3 + 0] - qx;
        float dy = X[i * 3 + 1] - qy;
        float dz = X[i * 3 + 2] - qz;
        const bool pred = (dx * dx + dy * dy + dz * dz) < R2;
        const unsigned mask = (unsigned)__ballot(pred);
        if (!found && mask) { first = n0 + __ffs((int)mask) - 1; found = true; }
        if (pred) {
            const int pos = cnt + __popc(mask & ((1u << lane) - 1u));
            if (pos < NSAMPLE) out[pos] = i;
        }
        cnt += __popc(mask);
        if (cnt >= NSAMPLE) break;
    }
    if (cnt < NSAMPLE) {
        const int f = found ? first : 0;
        if (lane >= cnt) out[lane] = f;
    }
}

// ---------------------------------------------------------------------------
// 3) Build features: [grouped_xyz - center | grouped_points | zero-pad] as
//    f16 rows of 96 (K padded to multiple of 32 for WMMA).
// ---------------------------------------------------------------------------
__global__ void build_feats_kernel(const float* __restrict__ xyz,
                                   const float* __restrict__ points,
                                   const int* __restrict__ idxb,
                                   const float* __restrict__ newxyz,
                                   _Float16* __restrict__ F)
{
    const int row  = blockIdx.x * 8 + (threadIdx.x >> 5);  // 524288 rows
    const int lane = threadIdx.x & 31;
    const int s = (row >> 5) & (NPOINT - 1);
    const int b = row >> 16;
    const int j = idxb[row];
    const float* xp  = xyz    + ((size_t)b * NPTS + j) * 3;
    const float* cp  = newxyz + ((size_t)b * NPOINT + s) * 3;
    const float* pp  = points + ((size_t)b * NPTS + j) * 64;
    _Float16* o = F + (size_t)row * 96;
    for (int e = lane; e < 96; e += 32) {
        float v;
        if      (e < 3)  v = xp[e] - cp[e];
        else if (e < 67) v = pp[e - 3];
        else             v = 0.0f;
        o[e] = (_Float16)v;
    }
}

// ---------------------------------------------------------------------------
// 4) Weight convert: f32 [Kreal x N] -> f16 [Kpad x N], zero-padded rows.
// ---------------------------------------------------------------------------
__global__ void cvt_w_kernel(const float* __restrict__ W, _Float16* __restrict__ Wh,
                             int Kreal, int Kpad, int N)
{
    const int g = blockIdx.x * 256 + threadIdx.x;
    if (g < Kpad * N) {
        const int k = g / N;
        Wh[g] = (k < Kreal) ? (_Float16)W[(size_t)k * N + (g % N)] : (_Float16)0.0f;
    }
}

// ---------------------------------------------------------------------------
// 5) WMMA GEMM, templated on (N, KPAD, BN) so the k-loop fully unrolls and
//    the BN+ReLU fusion is branch-free. Each wave owns one 16-row M tile and
//    computes ALL N/16 column tiles, reusing a single A fragment (A is read
//    exactly once per layer -> minimal HBM traffic). Weights staged in LDS.
//    Per-block column sums/sumsq written as deterministic partials.
// ---------------------------------------------------------------------------
template<int N, int KPAD, bool BN>
__global__ void gemm_bn_wmma_kernel(const _Float16* __restrict__ A,
                                    const _Float16* __restrict__ Wh,
                                    const float* __restrict__ scale,
                                    const float* __restrict__ shift,
                                    _Float16* __restrict__ Y,
                                    float* __restrict__ part)
{
    constexpr int TN = N / 16;                 // column tiles
    __shared__ uint4 lWraw[KPAD * N / 8];      // f16 weights, 16B chunks
    __shared__ float lws[8 * N];
    __shared__ float lwq[8 * N];
    __shared__ float lsc[64];
    __shared__ float lsh[64];

    const int tid = threadIdx.x;
    {   // stage weights to LDS (one pass, 16B per thread-iter)
        const uint4* src = (const uint4*)Wh;
        constexpr int NCHUNK = KPAD * N / 8;
        for (int i = tid; i < NCHUNK; i += 256) lWraw[i] = src[i];
    }
    if (BN && tid < 64) { lsc[tid] = scale[tid]; lsh[tid] = shift[tid]; }
    __syncthreads();
    const _Float16* lW = (const _Float16*)lWraw;

    const int wave = tid >> 5, lane = tid & 31;
    const int m0 = (blockIdx.x * 8 + wave) * 16;   // 128 rows per block
    const int m  = m0 + (lane & 15);
    const int hv = lane >> 4;                      // half-wave selector

    v8f acc[TN] = {};

    #pragma unroll
    for (int k0 = 0; k0 < KPAD; k0 += 32) {
        // --- A fragment: lane holds row m; halves cover K chunks
        //     [k0 + hv*8 .. +7] and [k0+16+hv*8 .. +7]  (ISA 16-bit A layout)
        const v8h c0 = *(const v8h*)(A + (size_t)m * KPAD + k0 + hv * 8);
        const v8h c1 = *(const v8h*)(A + (size_t)m * KPAD + k0 + 16 + hv * 8);
        v16h a;
        if (BN) {
            const int cb0 = k0 + hv * 8, cb1 = k0 + 16 + hv * 8;
            #pragma unroll
            for (int i = 0; i < 8; ++i) {
                float x0 = (float)c0[i] * lsc[cb0 + i] + lsh[cb0 + i];
                float x1 = (float)c1[i] * lsc[cb1 + i] + lsh[cb1 + i];
                a[i]     = (_Float16)(x0 > 0.0f ? x0 : 0.0f);
                a[i + 8] = (_Float16)(x1 > 0.0f ? x1 : 0.0f);
            }
        } else {
            #pragma unroll
            for (int i = 0; i < 8; ++i) { a[i] = c0[i]; a[i + 8] = c1[i]; }
        }
        // --- B fragments from LDS: lane = column (t*16 + lane%16), K chunks as A
        #pragma unroll
        for (int t = 0; t < TN; ++t) {
            const int nc = t * 16 + (lane & 15);
            v16h bf;
            #pragma unroll
            for (int i = 0; i < 8; ++i) { bf[i] = lW[(k0 + hv * 8 + i) * N + nc]; bf[i + 8] = lW[(k0 + 16 + hv * 8 + i) * N + nc]; }
            acc[t] = __builtin_amdgcn_wmma_f32_16x16x32_f16(
                         false, a, false, bf, (short)0, acc[t], false, false);
        }
    }

    // --- store D tiles (f16) + per-wave column stats (covers all N columns)
    #pragma unroll
    for (int t = 0; t < TN; ++t) {
        float s = 0.0f, q = 0.0f;
        #pragma unroll
        for (int r = 0; r < 8; ++r) {
            const float v = acc[t][r];
            Y[(size_t)(m0 + r + hv * 8) * N + t * 16 + (lane & 15)] = (_Float16)v;
            s += v; q += v * v;
        }
        s += __shfl_xor(s, 16, 32);
        q += __shfl_xor(q, 16, 32);
        if (lane < 16) {
            lws[wave * N + t * 16 + lane] = s;
            lwq[wave * N + t * 16 + lane] = q;
        }
    }
    __syncthreads();
    if (tid < N) {                             // fixed-order -> deterministic
        float ps = 0.0f, pq = 0.0f;
        #pragma unroll
        for (int w = 0; w < 8; ++w) { ps += lws[w * N + tid]; pq += lwq[w * N + tid]; }
        part[(size_t)blockIdx.x * (2 * N) + tid]     = ps;
        part[(size_t)blockIdx.x * (2 * N) + N + tid] = pq;
    }
}

// ---------------------------------------------------------------------------
// 6) Deterministic reduction of per-block partials -> [sum | sumsq] (2N).
// ---------------------------------------------------------------------------
__global__ void reduce_partials_kernel(const float* __restrict__ part,
                                       int nblocks, int stride,
                                       float* __restrict__ outsum)
{
    __shared__ float buf[256];
    const int c = blockIdx.x, tid = threadIdx.x;
    float s = 0.0f;
    for (int i = tid; i < nblocks; i += 256) s += part[(size_t)i * stride + c];
    buf[tid] = s; __syncthreads();
    for (int o = 128; o; o >>= 1) { if (tid < o) buf[tid] += buf[tid + o]; __syncthreads(); }
    if (tid == 0) outsum[c] = buf[0];
}

// ---------------------------------------------------------------------------
// 7) BN finalize: scale = g*rsqrt(var+eps), shift = b - mean*scale
// ---------------------------------------------------------------------------
__global__ void bn_finalize_kernel(const float* __restrict__ sumq,
                                   const float* __restrict__ g,
                                   const float* __restrict__ b,
                                   float* __restrict__ sc,
                                   float* __restrict__ sh, int C)
{
    const int c = threadIdx.x;
    if (c < C) {
        const float m = sumq[c] * BN_CNT_INV;
        const float v = sumq[C + c] * BN_CNT_INV - m * m;
        const float s = g[c] * rsqrtf(v + 1e-5f);
        sc[c] = s;
        sh[c] = b[c] - m * s;
    }
}

// ---------------------------------------------------------------------------
// 8) Apply BN3+ReLU and max over the 32 samples -> new_points (f32).
// ---------------------------------------------------------------------------
__global__ void bn_relu_maxpool_kernel(const _Float16* __restrict__ Y3,
                                       const float* __restrict__ sc,
                                       const float* __restrict__ sh,
                                       float* __restrict__ outp)
{
    const int g  = blockIdx.x * 256 + threadIdx.x;   // 2,097,152
    const int c  = g & 127;
    const int sg = g >> 7;                           // b*2048+s
    const size_t base = (size_t)sg * NSAMPLE * 128 + c;
    const float a = sc[c], d = sh[c];
    float m = 0.0f;                                  // relu >= 0
    for (int k = 0; k < NSAMPLE; ++k) {
        float v = (float)Y3[base + (size_t)k * 128] * a + d;
        v = v > 0.0f ? v : 0.0f;
        if (v > m) m = v;
    }
    outp[(size_t)sg * 128 + c] = m;
}

// ---------------------------------------------------------------------------
extern "C" void kernel_launch(void* const* d_in, const int* in_sizes, int n_in,
                              void* d_out, int out_size, void* d_ws, size_t ws_size,
                              hipStream_t stream)
{
    const float* xyz    = (const float*)d_in[0];
    const float* points = (const float*)d_in[1];
    const float* W1 = (const float*)d_in[2];
    const float* g1 = (const float*)d_in[3];
    const float* b1 = (const float*)d_in[4];
    const float* W2 = (const float*)d_in[5];
    const float* g2 = (const float*)d_in[6];
    const float* b2 = (const float*)d_in[7];
    const float* W3 = (const float*)d_in[8];
    const float* g3 = (const float*)d_in[9];
    const float* b3 = (const float*)d_in[10];

    float* out    = (float*)d_out;
    float* newxyz = out;                       // 8*2048*3  = 49152 floats
    float* newpts = out + NB * NPOINT * 3;     // 8*2048*128

    char* ws = (char*)d_ws;
    int*      fpsIdx = (int*)(ws + 0);                    //   64 KB
    int*      idxb   = (int*)(ws + 65536);                //    2 MB
    _Float16* w1h    = (_Float16*)(ws + 2162688);         // 96x64 f16
    _Float16* w2h    = (_Float16*)(ws + 2174976);         // 64x64 f16
    _Float16* w3h    = (_Float16*)(ws + 2183168);         // 64x128 f16
    float*    stats  = (float*)(ws + 2199552);            // 1024 f32
    float* sumq1 = stats;          // [sum64 | ssq64]
    float* sumq2 = stats + 128;
    float* sumq3 = stats + 256;    // [sum128 | ssq128]
    float* sc1 = stats + 512, *sh1 = stats + 576;
    float* sc2 = stats + 640, *sh2 = stats + 704;
    float* sc3 = stats + 768, *sh3 = stats + 896;
    _Float16* feats0 = (_Float16*)(ws + 4194304);         // 100 MB (reused as y2)
    _Float16* y1     = (_Float16*)(ws + 104857600);       // 134 MB region (reused as y3)
    _Float16* y2     = feats0;
    _Float16* y3     = y1;
    float*    part   = (float*)(ws + 239075328);          // <=4 MB partial sums

    const int GB = ROWS / 128;                 // 4096 gemm blocks (128 rows/blk)

    // stage 0: weights -> padded f16
    cvt_w_kernel<<<(96 * 64 + 255) / 256, 256, 0, stream>>>(W1, w1h, 67, 96, 64);
    cvt_w_kernel<<<(64 * 64 + 255) / 256, 256, 0, stream>>>(W2, w2h, 64, 64, 64);
    cvt_w_kernel<<<(64 * 128 + 255) / 256, 256, 0, stream>>>(W3, w3h, 64, 64, 128);

    // stage 1: sampling + grouping
    fps_kernel<<<NB, 1024, 0, stream>>>(xyz, fpsIdx, newxyz);
    ball_query_kernel<<<NB * NPOINT / 8, 256, 0, stream>>>(xyz, newxyz, idxb);
    build_feats_kernel<<<ROWS / 8, 256, 0, stream>>>(xyz, points, idxb, newxyz, feats0);

    // stage 2: MLP layer 1  (K=96 padded, N=64), no input BN
    gemm_bn_wmma_kernel<64, 96, false><<<GB, 256, 0, stream>>>(feats0, w1h,
                                                nullptr, nullptr, y1, part);
    reduce_partials_kernel<<<128, 256, 0, stream>>>(part, GB, 128, sumq1);
    bn_finalize_kernel<<<1, 64, 0, stream>>>(sumq1, g1, b1, sc1, sh1, 64);

    // stage 3: MLP layer 2  (BN1+ReLU fused into A load)
    gemm_bn_wmma_kernel<64, 64, true><<<GB, 256, 0, stream>>>(y1, w2h,
                                                sc1, sh1, y2, part);
    reduce_partials_kernel<<<128, 256, 0, stream>>>(part, GB, 128, sumq2);
    bn_finalize_kernel<<<1, 64, 0, stream>>>(sumq2, g2, b2, sc2, sh2, 64);

    // stage 4: MLP layer 3  (N=128, BN2+ReLU fused)
    gemm_bn_wmma_kernel<128, 64, true><<<GB, 256, 0, stream>>>(y2, w3h,
                                                sc2, sh2, y3, part);
    reduce_partials_kernel<<<256, 256, 0, stream>>>(part, GB, 256, sumq3);
    bn_finalize_kernel<<<1, 128, 0, stream>>>(sumq3, g3, b3, sc3, sh3, 128);

    // stage 5: BN3 + ReLU + max over samples
    bn_relu_maxpool_kernel<<<NB * NPOINT * 128 / 256, 256, 0, stream>>>(y3, sc3, sh3, newpts);
}
